// LMHA_49409303773804
// MI455X (gfx1250) — matrix operand order repaired and checked by reading, not compiled, not run on hardware
//
#include <hip/hip_runtime.h>
#include <hip/hip_bf16.h>

// ---------------------------------------------------------------------------
// LMHA + DLA for MI455X (gfx1250, wave32, WMMA).
// Pipeline (all on `stream`, graph-capture safe):
//   k_prep      : pack Wq, Wkv, Wp to f16 (row-major; B operands are staged
//                 k-contiguous = weight row-major, so no transposes)
//   k_zero      : zero GroupNorm stat accumulators
//   k_qproj     : q [B,C,N] -> qh f16 [B,NH,N,32]        (WMMA)
//   k_kvproj    : kv -> k f16 [B,NH,M,32], v^T f16 [B,NH,32,M]   (WMMA)
//   k_attn      : softmax(q k^T * s + rpb) -> fp32 attn  (WMMA, K=HD=32)
//   k_expand    : 1x1 expand 8->24, GN1 sums (atomics), x NOT stored
//   k_dwconv    : recompute expand+GN1+swish in LDS halo tile,
//                 depthwise 3x3, GN2 sums, y -> f16
//   k_reduce1x1 : GN2+swish, 1x1 reduce 24->8, GN3 sums, overwrite attn buf
//   k_av        : GN3 affine on the fly, attn @ v        (WMMA)
//   k_oproj     : out = oh @ Wp^T + bp -> [B,C,N]        (WMMA)
// Fragments: A and B staged k-contiguous -> each lane reads 2 contiguous 16B
// chunks -> ds_load_b128. Staging is vectorized: f16 tiles move as one
// global_load_b128 + ds_store_b128 per thread; fp32 tiles as float4 loads.
// Prefetches use locality 3 -> WGP scope (pull into all cache levels), since
// the same workgroup consumes the line one K-step later.
// Workspace: ~357 MB (attn fp32 134MB + y f16 201MB dominate).
// ---------------------------------------------------------------------------

typedef __attribute__((ext_vector_type(16))) _Float16 v16h;
typedef __attribute__((ext_vector_type(8)))  _Float16 v8h;
typedef __attribute__((ext_vector_type(4)))  _Float16 v4h;
typedef __attribute__((ext_vector_type(8)))  float    v8f;

static constexpr int kB = 16, kC = 256, kN = 1024, kM = 256;
static constexpr int kNH = 8, kHD = 32, kHID = 24;
static constexpr int kLd = 40;  // LDS row stride in halfs: 80B, 16B-aligned,
                                // bank step 20 -> conflict-free frag reads
static constexpr float kEPS = 1e-5f;
static constexpr float kScale = 0.17677669529663687f;  // HD^-0.5

static __device__ inline v8f wmma32(v16h a, v16h b, v8f c) {
  // emits v_wmma_f32_16x16x32_f16
  return __builtin_amdgcn_wmma_f32_16x16x32_f16(false, a, false, b,
                                                (short)0, c, false, false);
}
static __device__ inline v8f zero8() {
  v8f z = {0.f, 0.f, 0.f, 0.f, 0.f, 0.f, 0.f, 0.f};
  return z;
}

// A fragment 16x32 f16 from row-major [m][k] tile (k contiguous).
// per lane: halfs [8h..8h+7] and [16+8h..16+8h+7]: two b128 loads.
static __device__ inline v16h frag_a_kc(const _Float16* tile, int stride,
                                        int lane) {
  int m = lane & 15, h = lane >> 4;
  const _Float16* row = tile + m * stride;
  v8h c0 = *(const v8h*)(row + 8 * h);
  v8h c1 = *(const v8h*)(row + 16 + 8 * h);
  v16h f;
#pragma unroll
  for (int i = 0; i < 8; ++i) { f[i] = c0[i]; f[8 + i] = c1[i]; }
  return f;
}
// B fragment 32x16 f16 from B^T row-major [n][k] tile (k contiguous).
// per lane: halfs [8h..8h+7] and [16+8h..16+8h+7]: two b128 loads.
static __device__ inline v16h frag_b_kc(const _Float16* tile, int stride,
                                        int lane) {
  int n = lane & 15, h = lane >> 4;
  const _Float16* row = tile + n * stride;
  v8h c0 = *(const v8h*)(row + 8 * h);
  v8h c1 = *(const v8h*)(row + 16 + 8 * h);
  v16h f;
#pragma unroll
  for (int i = 0; i < 8; ++i) { f[i] = c0[i]; f[8 + i] = c1[i]; }
  return f;
}
static __device__ inline float wred(float v) {
#pragma unroll
  for (int o = 16; o > 0; o >>= 1) v += __shfl_xor(v, o, 32);
  return v;
}
static __device__ inline float swishf(float t) { return t / (1.f + __expf(-t)); }

// ---------------------------------------------------------------------------
__global__ __launch_bounds__(256) void k_prep(
    const float* __restrict__ Wq, const float* __restrict__ Wkv,
    const float* __restrict__ Wp, _Float16* __restrict__ Wq16,
    _Float16* __restrict__ Wkv16, _Float16* __restrict__ Wp16) {
  int i = blockIdx.x * 256 + threadIdx.x;
  if (i < kC * kC) {
    Wq16[i] = (_Float16)Wq[i];
    Wp16[i] = (_Float16)Wp[i];
  }
  if (i < 2 * kC * kC) Wkv16[i] = (_Float16)Wkv[i];
}

__global__ __launch_bounds__(256) void k_zero(float* __restrict__ st) {
  int i = blockIdx.x * 256 + threadIdx.x;
  if (i < 512) st[i] = 0.f;
}

// ---------------------------------------------------------------------------
// q projection: D[n][o] = sum_c q[b][c][n] * Wq[o][c].  64x64 block, 8 waves.
__global__ __launch_bounds__(256) void k_qproj(
    const float* __restrict__ q, const _Float16* __restrict__ Wq16,
    _Float16* __restrict__ qh) {
  __shared__ alignas(16) _Float16 Asm[64][kLd];  // [n][c]
  __shared__ alignas(16) _Float16 Bsm[64][kLd];  // [o][c] = Wq row-major
  const int b = blockIdx.z, n0 = blockIdx.x * 64, o0 = blockIdx.y * 64;
  const int tid = threadIdx.x, lane = tid & 31, wave = tid >> 5;
  const int wr = wave & 3, wc = wave >> 2;
  v8f acc0 = zero8(), acc1 = zero8();
  for (int k0 = 0; k0 < kC; k0 += 32) {
    // A: transpose q[b][c][n] -> Asm[n][c]; float4 over contiguous n.
#pragma unroll
    for (int i = 0; i < 2; ++i) {
      int idx = tid + i * 256;          // 0..511 = 32c x 16 n-groups
      int g = idx & 15, cl = idx >> 4;  // g: group of 4 n, cl: 0..31
      const float4 vq =
          *(const float4*)&q[((size_t)b * kC + k0 + cl) * kN + n0 + 4 * g];
      Asm[4 * g + 0][cl] = (_Float16)vq.x;
      Asm[4 * g + 1][cl] = (_Float16)vq.y;
      Asm[4 * g + 2][cl] = (_Float16)vq.z;
      Asm[4 * g + 3][cl] = (_Float16)vq.w;
    }
    // B: one b128 global load + one b128 LDS store per thread.
    {
      int ol = tid >> 2, cq = tid & 3;
      *(v8h*)&Bsm[ol][8 * cq] =
          *(const v8h*)&Wq16[(size_t)(o0 + ol) * kC + k0 + 8 * cq];
    }
    if (k0 + 32 < kC) {  // global_prefetch_b8, WGP scope (locality 3)
      __builtin_prefetch(&q[((size_t)b * kC + k0 + 32 + (tid >> 6)) * kN + n0 +
                            (tid & 63)], 0, 3);
      __builtin_prefetch(&Wq16[(size_t)(o0 + (tid >> 2)) * kC + k0 + 32 +
                               8 * (tid & 3)], 0, 3);
    }
    __syncthreads();
    v16h a  = frag_a_kc(&Asm[wr * 16][0], kLd, lane);
    v16h b0 = frag_b_kc(&Bsm[wc * 32][0], kLd, lane);
    v16h b1 = frag_b_kc(&Bsm[wc * 32 + 16][0], kLd, lane);
    acc0 = wmma32(a, b0, acc0);
    acc1 = wmma32(a, b1, acc1);
    __syncthreads();
  }
  int nn = lane & 15, hl = lane >> 4;
#pragma unroll
  for (int r = 0; r < 8; ++r) {
    int row = n0 + wr * 16 + r + 8 * hl;
#pragma unroll
    for (int t = 0; t < 2; ++t) {
      int o = o0 + wc * 32 + t * 16 + nn;
      float val = t ? acc1[r] : acc0[r];
      int hh = o >> 5, dd = o & 31;
      qh[(((size_t)b * kNH + hh) * kN + row) * kHD + dd] = (_Float16)val;
    }
  }
}

// kv projection -> k f16 [B,NH,M,HD] and v^T f16 [B,NH,HD,M]
__global__ __launch_bounds__(256) void k_kvproj(
    const float* __restrict__ kv, const _Float16* __restrict__ Wkv16,
    _Float16* __restrict__ kk, _Float16* __restrict__ vT) {
  __shared__ alignas(16) _Float16 Asm[64][kLd];
  __shared__ alignas(16) _Float16 Bsm[64][kLd];
  const int b = blockIdx.z, m0 = blockIdx.x * 64, o0 = blockIdx.y * 64;
  const int tid = threadIdx.x, lane = tid & 31, wave = tid >> 5;
  const int wr = wave & 3, wc = wave >> 2;
  v8f acc0 = zero8(), acc1 = zero8();
  for (int k0 = 0; k0 < kC; k0 += 32) {
#pragma unroll
    for (int i = 0; i < 2; ++i) {
      int idx = tid + i * 256;
      int g = idx & 15, cl = idx >> 4;
      const float4 vk =
          *(const float4*)&kv[((size_t)b * kC + k0 + cl) * kM + m0 + 4 * g];
      Asm[4 * g + 0][cl] = (_Float16)vk.x;
      Asm[4 * g + 1][cl] = (_Float16)vk.y;
      Asm[4 * g + 2][cl] = (_Float16)vk.z;
      Asm[4 * g + 3][cl] = (_Float16)vk.w;
    }
    {
      int ol = tid >> 2, cq = tid & 3;
      *(v8h*)&Bsm[ol][8 * cq] =
          *(const v8h*)&Wkv16[(size_t)(o0 + ol) * kC + k0 + 8 * cq];
    }
    if (k0 + 32 < kC) {
      __builtin_prefetch(&kv[((size_t)b * kC + k0 + 32 + (tid >> 6)) * kM + m0 +
                             (tid & 63)], 0, 3);
      __builtin_prefetch(&Wkv16[(size_t)(o0 + (tid >> 2)) * kC + k0 + 32 +
                                8 * (tid & 3)], 0, 3);
    }
    __syncthreads();
    v16h a  = frag_a_kc(&Asm[wr * 16][0], kLd, lane);
    v16h b0 = frag_b_kc(&Bsm[wc * 32][0], kLd, lane);
    v16h b1 = frag_b_kc(&Bsm[wc * 32 + 16][0], kLd, lane);
    acc0 = wmma32(a, b0, acc0);
    acc1 = wmma32(a, b1, acc1);
    __syncthreads();
  }
  int nn = lane & 15, hl = lane >> 4;
#pragma unroll
  for (int r = 0; r < 8; ++r) {
    int mrow = m0 + wr * 16 + r + 8 * hl;
#pragma unroll
    for (int t = 0; t < 2; ++t) {
      int o = o0 + wc * 32 + t * 16 + nn;
      float val = t ? acc1[r] : acc0[r];
      if (o < kC) {  // k: row-major [m][d] (feeds QK^T B-tile directly)
        int hh = o >> 5, dd = o & 31;
        kk[(((size_t)b * kNH + hh) * kM + mrow) * kHD + dd] = (_Float16)val;
      } else {       // v: transposed [d][m] (feeds AV B-tile directly)
        int o2 = o - kC, hh = o2 >> 5, dd = o2 & 31;
        vT[(((size_t)b * kNH + hh) * kHD + dd) * kM + mrow] = (_Float16)val;
      }
    }
  }
}

// ---------------------------------------------------------------------------
// Fused QK^T + bias + softmax. Block = 16 query rows x full 256 keys, 8 waves.
__global__ __launch_bounds__(256) void k_attn(
    const _Float16* __restrict__ qh, const _Float16* __restrict__ kk,
    const float* __restrict__ rpb, float* __restrict__ attn) {
  __shared__ alignas(16) _Float16 Asm[16][kLd];   // q tile [n][d]
  __shared__ alignas(16) _Float16 Ksm[256][kLd];  // B^T = k row-major [m][d]
  __shared__ float Ssm[16][256];
  const int b = blockIdx.z, h = blockIdx.y, n0 = blockIdx.x * 16;
  const int tid = threadIdx.x, lane = tid & 31, wave = tid >> 5;
  if (tid < 64) {  // 16x32 halfs = 64 v8h chunks
    int r = tid >> 2, cq = tid & 3;
    *(v8h*)&Asm[r][8 * cq] =
        *(const v8h*)&qh[(((size_t)b * kNH + h) * kN + n0 + r) * kHD + 8 * cq];
  }
#pragma unroll
  for (int i = 0; i < 4; ++i) {  // 256x32 halfs = 1024 v8h chunks
    int idx = tid + i * 256;
    int m = idx >> 2, cq = idx & 3;
    *(v8h*)&Ksm[m][8 * cq] =
        *(const v8h*)&kk[(((size_t)b * kNH + h) * kM + m) * kHD + 8 * cq];
  }
  __syncthreads();
  const int m0 = wave * 32;
  v16h a  = frag_a_kc(&Asm[0][0], kLd, lane);
  v16h b0 = frag_b_kc(&Ksm[m0][0], kLd, lane);
  v16h b1 = frag_b_kc(&Ksm[m0 + 16][0], kLd, lane);
  v8f c0 = zero8(), c1 = zero8();
  c0 = wmma32(a, b0, c0);  // single K=32 step (= HD)
  c1 = wmma32(a, b1, c1);
  int nn = lane & 15, hl = lane >> 4;
#pragma unroll
  for (int r = 0; r < 8; ++r) {
    Ssm[r + 8 * hl][m0 + nn]      = c0[r];
    Ssm[r + 8 * hl][m0 + 16 + nn] = c1[r];
  }
  __syncthreads();
  // softmax: 16 threads per row; the 16-thread group lives in one shfl width.
  const int row = tid >> 4, j = tid & 15;
  float vals[16];
  float vmax = -3.4e38f;
#pragma unroll
  for (int i = 0; i < 16; ++i) {
    int m = j + 16 * i;
    float v = Ssm[row][m] * kScale + rpb[(size_t)(n0 + row) * kM + m];
    vals[i] = v;
    vmax = fmaxf(vmax, v);
  }
#pragma unroll
  for (int o = 8; o > 0; o >>= 1) vmax = fmaxf(vmax, __shfl_xor(vmax, o, 16));
  float sum = 0.f;
#pragma unroll
  for (int i = 0; i < 16; ++i) { vals[i] = __expf(vals[i] - vmax); sum += vals[i]; }
#pragma unroll
  for (int o = 8; o > 0; o >>= 1) sum += __shfl_xor(sum, o, 16);
  float inv = 1.f / sum;
#pragma unroll
  for (int i = 0; i < 16; ++i)
    attn[(((size_t)b * kNH + h) * kN + n0 + row) * kM + j + 16 * i] = vals[i] * inv;
}

// ---------------------------------------------------------------------------
// 1x1 expand (8->24) + GN1 group sums. x is recomputed later, not stored.
__global__ __launch_bounds__(256) void k_expand(
    const float* __restrict__ attn, const float* __restrict__ Wexp,
    float* __restrict__ st1) {
  __shared__ float We[192];
  __shared__ float red[8][6];
  const int b = blockIdx.y, tid = threadIdx.x;
  if (tid < 192) We[tid] = Wexp[tid];
  __syncthreads();
  const size_t NM = (size_t)kN * kM;
  size_t p = (size_t)blockIdx.x * 256 + tid;
  float a[8];
#pragma unroll
  for (int c = 0; c < 8; ++c) a[c] = attn[((size_t)b * kNH + c) * NM + p];
  float gs[3] = {0, 0, 0}, gq[3] = {0, 0, 0};
#pragma unroll
  for (int o = 0; o < kHID; ++o) {
    float x = 0.f;
#pragma unroll
    for (int c = 0; c < 8; ++c) x += We[o * 8 + c] * a[c];
    gs[o >> 3] += x;
    gq[o >> 3] += x * x;
  }
  int lane = tid & 31, wave = tid >> 5;
#pragma unroll
  for (int g = 0; g < 3; ++g) {
    float s = wred(gs[g]), qv = wred(gq[g]);
    if (lane == 0) { red[wave][g * 2] = s; red[wave][g * 2 + 1] = qv; }
  }
  __syncthreads();
  if (tid < 6) {
    float s = 0.f;
    for (int w = 0; w < 8; ++w) s += red[w][tid];
    atomicAdd(&st1[b * 6 + tid], s);
  }
}

// GN1-affine + swish recomputed into LDS halo tile, depthwise 3x3, GN2 sums.
__global__ __launch_bounds__(256) void k_dwconv(
    const float* __restrict__ attn, const float* __restrict__ Wexp,
    const float* __restrict__ gamma1, const float* __restrict__ beta1,
    const float* __restrict__ Wdw, const float* __restrict__ st1,
    float* __restrict__ st2, _Float16* __restrict__ y) {
  __shared__ float s[kHID][10][34];  // (8+2)x(32+2) halo tile per channel
  __shared__ float We[192];
  __shared__ float Wd[216];
  __shared__ float aa[kHID], ab[kHID];
  __shared__ float red[8][6];
  const int b = blockIdx.z, n0 = blockIdx.y * 8, m0 = blockIdx.x * 32;
  const int tid = threadIdx.x;
  if (tid < 192) We[tid] = Wexp[tid];
  if (tid < 216) Wd[tid] = Wdw[tid];
  if (tid < kHID) {
    int g = tid >> 3;
    float cnt = 8.f * kN * kM;
    float mu  = st1[b * 6 + g * 2] / cnt;
    float var = st1[b * 6 + g * 2 + 1] / cnt - mu * mu;
    float rs  = rsqrtf(var + kEPS);
    aa[tid] = rs * gamma1[tid];
    ab[tid] = beta1[tid] - mu * rs * gamma1[tid];
  }
  __syncthreads();
  for (int p = tid; p < 10 * 34; p += 256) {
    int py = p / 34, px = p % 34;
    int n = n0 + py - 1, m = m0 + px - 1;
    bool in = (n >= 0) && (n < kN) && (m >= 0) && (m < kM);
    float a[8];
    if (in) {
#pragma unroll
      for (int c = 0; c < 8; ++c)
        a[c] = attn[(((size_t)b * kNH + c) * kN + n) * kM + m];
    }
#pragma unroll
    for (int o = 0; o < kHID; ++o) {
      float v = 0.f;  // conv padding = zero in swish(gn(x)) space
      if (in) {
        float x = 0.f;
#pragma unroll
        for (int c = 0; c < 8; ++c) x += We[o * 8 + c] * a[c];
        v = swishf(x * aa[o] + ab[o]);
      }
      s[o][py][px] = v;
    }
  }
  __syncthreads();
  float gs[3] = {0, 0, 0}, gq[3] = {0, 0, 0};
#pragma unroll
  for (int c = 0; c < kHID; ++c) {  // channel per iteration, pixel = tid
    int py = tid >> 5, px = tid & 31;
    float acc = 0.f;
#pragma unroll
    for (int i = 0; i < 3; ++i)
#pragma unroll
      for (int j = 0; j < 3; ++j)
        acc += Wd[c * 9 + i * 3 + j] * s[c][py + i][px + j];
    gs[c >> 3] += acc;
    gq[c >> 3] += acc * acc;
    y[(((size_t)b * kHID + c) * kN + (n0 + py)) * kM + (m0 + px)] = (_Float16)acc;
  }
  int lane = tid & 31, wave = tid >> 5;
#pragma unroll
  for (int g = 0; g < 3; ++g) {
    float sv = wred(gs[g]), qv = wred(gq[g]);
    if (lane == 0) { red[wave][g * 2] = sv; red[wave][g * 2 + 1] = qv; }
  }
  __syncthreads();
  if (tid < 6) {
    float sum = 0.f;
    for (int w = 0; w < 8; ++w) sum += red[w][tid];
    atomicAdd(&st2[b * 6 + tid], sum);
  }
}

// GN2-affine + swish + 1x1 reduce (24->8) + GN3 sums; overwrites attn buffer.
__global__ __launch_bounds__(256) void k_reduce1x1(
    const _Float16* __restrict__ y, const float* __restrict__ Wred,
    const float* __restrict__ gamma2, const float* __restrict__ beta2,
    const float* __restrict__ st2, float* __restrict__ st3,
    float* __restrict__ a2) {
  __shared__ float Wr[192];
  __shared__ float aa[kHID], ab[kHID];
  __shared__ float red[8][2];
  const int b = blockIdx.y, tid = threadIdx.x;
  if (tid < 192) Wr[tid] = Wred[tid];
  if (tid < kHID) {
    int g = tid >> 3;
    float cnt = 8.f * kN * kM;
    float mu  = st2[b * 6 + g * 2] / cnt;
    float var = st2[b * 6 + g * 2 + 1] / cnt - mu * mu;
    float rs  = rsqrtf(var + kEPS);
    aa[tid] = rs * gamma2[tid];
    ab[tid] = beta2[tid] - mu * rs * gamma2[tid];
  }
  __syncthreads();
  const size_t NM = (size_t)kN * kM;
  size_t p = (size_t)blockIdx.x * 256 + tid;
  float s2[kHID];
#pragma unroll
  for (int c = 0; c < kHID; ++c) {
    float yv = (float)y[((size_t)b * kHID + c) * NM + p];
    s2[c] = swishf(yv * aa[c] + ab[c]);
  }
  float ls = 0.f, lq = 0.f;
#pragma unroll
  for (int o = 0; o < kNH; ++o) {
    float v = 0.f;
#pragma unroll
    for (int c = 0; c < kHID; ++c) v += Wr[o * kHID + c] * s2[c];
    a2[((size_t)b * kNH + o) * NM + p] = v;
    ls += v;
    lq += v * v;
  }
  int lane = tid & 31, wave = tid >> 5;
  ls = wred(ls);
  lq = wred(lq);
  if (lane == 0) { red[wave][0] = ls; red[wave][1] = lq; }
  __syncthreads();
  if (tid < 2) {
    float sum = 0.f;
    for (int w = 0; w < 8; ++w) sum += red[w][tid];
    atomicAdd(&st3[b * 2 + tid], sum);
  }
}

// ---------------------------------------------------------------------------
// out_heads = gn3(a2) @ v : per (b,h) 1024x32 GEMM, K=256. 64-row block.
__global__ __launch_bounds__(256) void k_av(
    const float* __restrict__ a2, const _Float16* __restrict__ vT,
    const float* __restrict__ gamma3, const float* __restrict__ beta3,
    const float* __restrict__ st3, _Float16* __restrict__ oh) {
  __shared__ alignas(16) _Float16 Asm[64][kLd];  // [n][m]
  __shared__ alignas(16) _Float16 Bsm[32][kLd];  // B^T = vT row-major [d][m]
  const int b = blockIdx.z, h = blockIdx.y, n0 = blockIdx.x * 64;
  const int tid = threadIdx.x, lane = tid & 31, wave = tid >> 5;
  const int wr = wave & 3, wc = wave >> 2;
  float cnt = 8.f * kN * kM;
  float mu  = st3[b * 2] / cnt;
  float var = st3[b * 2 + 1] / cnt - mu * mu;
  float rs  = rsqrtf(var + kEPS);
  float aaf = rs * gamma3[h];
  float abf = beta3[h] - mu * rs * gamma3[h];
  v8f acc = zero8();
  for (int k0 = 0; k0 < kM; k0 += 32) {
    // A: float4 over contiguous k, GN3 affine folded in, packed v4h store.
#pragma unroll
    for (int i = 0; i < 2; ++i) {
      int idx = tid + i * 256;         // 0..511 = 64 rows x 8 k-quads
      int kq = idx & 7, r = idx >> 3;
      const float4 va = *(const float4*)&a2[(((size_t)b * kNH + h) * kN + n0 +
                                             r) * kM + k0 + 4 * kq];
      v4h hv;
      hv[0] = (_Float16)(va.x * aaf + abf);
      hv[1] = (_Float16)(va.y * aaf + abf);
      hv[2] = (_Float16)(va.z * aaf + abf);
      hv[3] = (_Float16)(va.w * aaf + abf);
      *(v4h*)&Asm[r][4 * kq] = hv;
    }
    if (tid < 128) {  // B: 32x32 halfs = 128 v8h chunks
      int dd = tid >> 2, cq = tid & 3;
      *(v8h*)&Bsm[dd][8 * cq] = *(const v8h*)&vT[(((size_t)b * kNH + h) * kHD +
                                                  dd) * kM + k0 + 8 * cq];
    }
    if (k0 + 32 < kM) {
      __builtin_prefetch(&a2[(((size_t)b * kNH + h) * kN + n0 + (tid >> 5)) *
                                 kM + k0 + 32 + (tid & 31)], 0, 3);
    }
    __syncthreads();
    v16h a  = frag_a_kc(&Asm[wr * 16][0], kLd, lane);
    v16h bb = frag_b_kc(&Bsm[wc * 16][0], kLd, lane);
    acc = wmma32(a, bb, acc);
    __syncthreads();
  }
  int nn = lane & 15, hl = lane >> 4;
#pragma unroll
  for (int r = 0; r < 8; ++r) {
    int row = n0 + wr * 16 + r + 8 * hl;
    int d = wc * 16 + nn;
    oh[((size_t)b * kN + row) * kC + h * kHD + d] = (_Float16)acc[r];
  }
}

// out[b][o][n] = sum_c Wp[o][c]*oh[b][n][c] + bp[o]. Rows=o so n is contiguous.
__global__ __launch_bounds__(256) void k_oproj(
    const _Float16* __restrict__ oh, const _Float16* __restrict__ Wp16,
    const float* __restrict__ bp, float* __restrict__ out) {
  __shared__ alignas(16) _Float16 Asm[64][kLd];  // [o][c] = Wp row-major
  __shared__ alignas(16) _Float16 Bsm[64][kLd];  // B^T = oh row-major [n][c]
  const int b = blockIdx.z, o0 = blockIdx.y * 64, n0 = blockIdx.x * 64;
  const int tid = threadIdx.x, lane = tid & 31, wave = tid >> 5;
  const int wr = wave & 3, wc = wave >> 2;
  v8f acc0 = zero8(), acc1 = zero8();
  for (int k0 = 0; k0 < kC; k0 += 32) {
    {  // A: Wp row-major, one v8h per thread
      int r = tid >> 2, cq = tid & 3;
      *(v8h*)&Asm[r][8 * cq] =
          *(const v8h*)&Wp16[(size_t)(o0 + r) * kC + k0 + 8 * cq];
    }
    {  // B: oh row-major, one v8h per thread
      int nl = tid >> 2, cq = tid & 3;
      *(v8h*)&Bsm[nl][8 * cq] =
          *(const v8h*)&oh[((size_t)b * kN + n0 + nl) * kC + k0 + 8 * cq];
    }
    if (k0 + 32 < kC) {
      __builtin_prefetch(&Wp16[(size_t)(o0 + (tid >> 2)) * kC + k0 + 32 +
                               8 * (tid & 3)], 0, 3);
      __builtin_prefetch(&oh[((size_t)b * kN + n0 + (tid >> 2)) * kC + k0 + 32 +
                             8 * (tid & 3)], 0, 3);
    }
    __syncthreads();
    v16h a  = frag_a_kc(&Asm[wr * 16][0], kLd, lane);
    v16h b0 = frag_b_kc(&Bsm[wc * 32][0], kLd, lane);
    v16h b1 = frag_b_kc(&Bsm[wc * 32 + 16][0], kLd, lane);
    acc0 = wmma32(a, b0, acc0);
    acc1 = wmma32(a, b1, acc1);
    __syncthreads();
  }
  int nn = lane & 15, hl = lane >> 4;
#pragma unroll
  for (int r = 0; r < 8; ++r) {
    int orow = o0 + wr * 16 + r + 8 * hl;
    float bias = bp[orow];
    int col0 = n0 + wc * 32 + nn;
    out[((size_t)b * kC + orow) * kN + col0]      = acc0[r] + bias;
    out[((size_t)b * kC + orow) * kN + col0 + 16] = acc1[r] + bias;
  }
}

// ---------------------------------------------------------------------------
extern "C" void kernel_launch(void* const* d_in, const int* in_sizes, int n_in,
                              void* d_out, int out_size, void* d_ws,
                              size_t ws_size, hipStream_t stream) {
  (void)in_sizes; (void)n_in; (void)out_size; (void)ws_size;
  const float* q    = (const float*)d_in[0];
  const float* kv   = (const float*)d_in[1];
  const float* Wq   = (const float*)d_in[2];
  const float* Wkv  = (const float*)d_in[3];
  const float* Wp   = (const float*)d_in[4];
  const float* bp   = (const float*)d_in[5];
  const float* rpb  = (const float*)d_in[6];
  const float* Wexp = (const float*)d_in[7];
  const float* g1   = (const float*)d_in[8];
  const float* b1   = (const float*)d_in[9];
  const float* Wdw  = (const float*)d_in[10];
  const float* g2   = (const float*)d_in[11];
  const float* b2   = (const float*)d_in[12];
  const float* Wred = (const float*)d_in[13];
  const float* g3   = (const float*)d_in[14];
  const float* b3   = (const float*)d_in[15];
  float* out = (float*)d_out;

  char* ws = (char*)d_ws;
  size_t off = 0;
  auto take = [&](size_t bytes) {
    char* p = ws + off;
    off = (off + bytes + 255) & ~(size_t)255;
    return p;
  };
  _Float16* Wq16  = (_Float16*)take((size_t)kC * kC * 2);
  _Float16* Wkv16 = (_Float16*)take((size_t)kC * 2 * kC * 2);
  _Float16* Wp16  = (_Float16*)take((size_t)kC * kC * 2);
  _Float16* qh    = (_Float16*)take((size_t)kB * kNH * kN * kHD * 2);
  _Float16* k16   = (_Float16*)take((size_t)kB * kNH * kM * kHD * 2);
  _Float16* vT16  = (_Float16*)take((size_t)kB * kNH * kM * kHD * 2);
  _Float16* oh    = (_Float16*)take((size_t)kB * kN * kC * 2);
  float*    stats = (float*)take(4096);
  float* st1 = stats;        // [B][3][2]
  float* st2 = stats + 128;  // [B][3][2]
  float* st3 = stats + 256;  // [B][2]
  float*    attn  = (float*)take((size_t)kB * kNH * kN * kM * 4);
  _Float16* y16   = (_Float16*)take((size_t)kB * kHID * kN * kM * 2);

  k_prep<<<dim3((2 * kC * kC + 255) / 256), 256, 0, stream>>>(Wq, Wkv, Wp, Wq16,
                                                              Wkv16, Wp16);
  k_zero<<<2, 256, 0, stream>>>(stats);
  k_qproj<<<dim3(kN / 64, kC / 64, kB), 256, 0, stream>>>(q, Wq16, qh);
  k_kvproj<<<dim3(kM / 64, 2 * kC / 64, kB), 256, 0, stream>>>(kv, Wkv16, k16,
                                                               vT16);
  k_attn<<<dim3(kN / 16, kNH, kB), 256, 0, stream>>>(qh, k16, rpb, attn);
  k_expand<<<dim3(kN * kM / 256, kB), 256, 0, stream>>>(attn, Wexp, st1);
  k_dwconv<<<dim3(kM / 32, kN / 8, kB), 256, 0, stream>>>(attn, Wexp, g1, b1,
                                                          Wdw, st1, st2, y16);
  k_reduce1x1<<<dim3(kN * kM / 256, kB), 256, 0, stream>>>(y16, Wred, g2, b2,
                                                           st2, st3, attn);
  k_av<<<dim3(kN / 64, kNH, kB), 256, 0, stream>>>(attn, vT16, g3, b3, st3, oh);
  k_oproj<<<dim3(kN / 64, kC / 64, kB), 256, 0, stream>>>(oh, Wp16, bp, out);
}
